// MyLayerNormLSTM_11845519803060
// MI455X (gfx1250) — matrix operand order repaired and checked by reading, not compiled, not run on hardware
//
#include <hip/hip_runtime.h>
#include <hip/hip_bf16.h>

#define LN_EPS 1e-5f

typedef __attribute__((ext_vector_type(16))) __bf16 v16bf;
typedef __attribute__((ext_vector_type(8)))  __bf16 v8bf;
typedef __attribute__((ext_vector_type(8)))  float  v8f;

__device__ __forceinline__ float sigmoidf_(float x) { return 1.0f / (1.0f + expf(-x)); }

// ---------------------------------------------------------------------------
// Joint LayerNorm over (S,D)=12800 per batch row; output bf16 (B,S,D)
// ---------------------------------------------------------------------------
__global__ void __launch_bounds__(256) ln_input_kernel(
    const float* __restrict__ x, const float* __restrict__ g,
    const float* __restrict__ b, __bf16* __restrict__ xn)
{
    const int N = 200 * 64;
    int row = blockIdx.x;
    const float* xr = x + (long)row * N;
    float s = 0.f, ss = 0.f;
    for (int i = threadIdx.x; i < N; i += 256) { float v = xr[i]; s += v; ss += v * v; }
    __shared__ float sh[256], sh2[256];
    sh[threadIdx.x] = s; sh2[threadIdx.x] = ss;
    __syncthreads();
    for (int off = 128; off > 0; off >>= 1) {
        if (threadIdx.x < off) {
            sh[threadIdx.x]  += sh[threadIdx.x + off];
            sh2[threadIdx.x] += sh2[threadIdx.x + off];
        }
        __syncthreads();
    }
    float mean = sh[0] / N;
    float var  = sh2[0] / N - mean * mean;
    float rs   = rsqrtf(var + LN_EPS);
    __bf16* xo = xn + (long)row * N;
    for (int i = threadIdx.x; i < N; i += 256)
        xo[i] = (__bf16)((xr[i] - mean) * rs * g[i] + b[i]);
}

// ---------------------------------------------------------------------------
// One-shot fp32 -> bf16 conversion of all four recurrent weight matrices.
// Region sizes: Wih0 2048*64, then Whh0 / Wih1 / Whh1 each 2048*512.
// ---------------------------------------------------------------------------
__global__ void cvt_weights_kernel(
    const float* __restrict__ w0, const float* __restrict__ w1,
    const float* __restrict__ w2, const float* __restrict__ w3,
    __bf16* __restrict__ o0, __bf16* __restrict__ o1,
    __bf16* __restrict__ o2, __bf16* __restrict__ o3)
{
    const int N0 = 2048 * 64, N1 = 2048 * 512;
    long i = (long)blockIdx.x * blockDim.x + threadIdx.x;
    if (i < N0)              { o0[i] = (__bf16)w0[i]; return; }
    i -= N0;
    if (i < N1)              { o1[i] = (__bf16)w1[i]; return; }
    i -= N1;
    if (i < N1)              { o2[i] = (__bf16)w2[i]; return; }
    i -= N1;
    if (i < N1)              { o3[i] = (__bf16)w3[i]; }
}

__global__ void zero_kernel(float* __restrict__ p, int n)
{
    int i = blockIdx.x * blockDim.x + threadIdx.x;
    if (i < n) p[i] = 0.f;
}

// ---------------------------------------------------------------------------
// Dual WMMA bf16 GEMM: two independent problems in one launch.
//   problem p: Cp[m,n] = sum_k Ap[m,k]*Wp[n,k] + biasp[n]
// Fixed M=512, N=2048, ldc=2048. Blocks 0..31 -> problem 0, 32..63 -> problem 1.
// One wave computes a 64x64 tile: 4x4 fragments of v_wmma_f32_16x16x32_bf16.
// Fragment layouts per CDNA5 ISA 7.12.2:
//   A (16x32): lanes 0-15 -> M=lane, K=[0..8)+[16..24); lanes 16-31 -> K=[8..16)+[24..32)
//   B (32x16): lanes 0-15 -> N=lane, K=[0..8)+[16..24); lanes 16-31 -> K=[8..16)+[24..32)
//   D (16x16 f32): vgpr r -> M = r + 8*(lane>=16); N = lane%16
// ---------------------------------------------------------------------------
__global__ void __launch_bounds__(256) gemm_dual_kernel(
    const __bf16* __restrict__ A0, long lda0,
    const __bf16* __restrict__ W0, long ldw0,
    const float* __restrict__ bias0, float* __restrict__ C0, int K0,
    const __bf16* __restrict__ A1, long lda1,
    const __bf16* __restrict__ W1, long ldw1,
    const float* __restrict__ bias1, float* __restrict__ C1, int K1)
{
    const int NT = 2048 >> 6;           // 32 n-tiles
    const long ldc = 2048;

    bool second = (blockIdx.x >= 32);
    int  bx     = second ? (int)blockIdx.x - 32 : (int)blockIdx.x;
    const __bf16* A    = second ? A1 : A0;
    const __bf16* W    = second ? W1 : W0;
    const float*  bias = second ? bias1 : bias0;
    float*        C    = second ? C1 : C0;
    long lda = second ? lda1 : lda0;
    long ldw = second ? ldw1 : ldw0;
    int  K   = second ? K1 : K0;

    int wave = bx * 8 + ((int)threadIdx.x >> 5);
    int lane = threadIdx.x & 31;
    int mt = wave / NT;
    int nt = wave - mt * NT;
    int m0 = mt << 6, n0 = nt << 6;

    int half = lane >> 4;   // 0: lanes 0-15, 1: lanes 16-31
    int l16  = lane & 15;

    v8f acc[4][4] = {};

    for (int k0 = 0; k0 < K; k0 += 32) {
        v16bf afrag[4], bfrag[4];
#pragma unroll
        for (int i = 0; i < 4; i++) {
            // A fragment: two 16B chunks at K = k0 + half*8 and +16 further
            const __bf16* ap = A + (long)(m0 + i * 16 + l16) * lda + k0 + half * 8;
            v8bf alo = *(const v8bf*)ap;
            v8bf ahi = *(const v8bf*)(ap + 16);
            afrag[i] = __builtin_shufflevector(alo, ahi,
                        0,1,2,3,4,5,6,7,8,9,10,11,12,13,14,15);
            // B fragment: two 16B chunks at K = k0 + half*8 and +16 further
            const __bf16* bp = W + (long)(n0 + i * 16 + l16) * ldw + k0 + half * 8;
            v8bf blo = *(const v8bf*)bp;
            v8bf bhi = *(const v8bf*)(bp + 16);
            bfrag[i] = __builtin_shufflevector(blo, bhi,
                        0,1,2,3,4,5,6,7,8,9,10,11,12,13,14,15);
        }
#pragma unroll
        for (int i = 0; i < 4; i++)
#pragma unroll
            for (int j = 0; j < 4; j++)
                acc[i][j] = __builtin_amdgcn_wmma_f32_16x16x32_bf16(
                    false, afrag[i], false, bfrag[j],
                    (short)0, acc[i][j], false, false);
    }

#pragma unroll
    for (int i = 0; i < 4; i++) {
#pragma unroll
        for (int j = 0; j < 4; j++) {
            int col = n0 + j * 16 + l16;
            float bv = bias[col];
#pragma unroll
            for (int r = 0; r < 8; r++) {
                int row = m0 + i * 16 + half * 8 + r;
                C[(long)row * ldc + col] = acc[i][j][r] + bv;
            }
        }
    }
}

// ---------------------------------------------------------------------------
// Fused LayerNorm-LSTM cell. One block per batch row, 512 threads (= H).
// Thread tid handles gate columns tid, tid+512, tid+1024, tid+1536 = i,f,g,o.
//   gates = LN_2048(Gih)*g_ih+be_ih + LN_2048(Ghh)*g_hh+be_hh
//   c'    = sig(f)*c + sig(i)*tanh(g)
//   h'    = sig(o) * (LN_512(tanh(c'))*g_ho + be_ho)
// ---------------------------------------------------------------------------
__global__ void __launch_bounds__(512) lstm_cell_kernel(
    const float* __restrict__ Gih, const float* __restrict__ Ghh,
    const float* __restrict__ g_ih, const float* __restrict__ be_ih,
    const float* __restrict__ g_hh, const float* __restrict__ be_hh,
    const float* __restrict__ g_ho, const float* __restrict__ be_ho,
    float* __restrict__ c, __bf16* __restrict__ h)
{
    const int NG = 2048, H = 512;
    int b = blockIdx.x, tid = threadIdx.x;
    const float* gi = Gih + (long)b * NG;
    const float* gh = Ghh + (long)b * NG;

    float iv[4], hv[4];
    float si = 0.f, qi = 0.f, sh0 = 0.f, qh = 0.f;
#pragma unroll
    for (int q = 0; q < 4; q++) {
        float v = gi[tid + H * q]; iv[q] = v; si += v; qi += v * v;
        float w = gh[tid + H * q]; hv[q] = w; sh0 += w; qh += w * w;
    }

    __shared__ float4 red[512];
    red[tid] = make_float4(si, qi, sh0, qh);
    __syncthreads();
    for (int off = 256; off > 0; off >>= 1) {
        if (tid < off) {
            float4 a = red[tid], d = red[tid + off];
            red[tid] = make_float4(a.x + d.x, a.y + d.y, a.z + d.z, a.w + d.w);
        }
        __syncthreads();
    }
    float4 tot = red[0];
    __syncthreads();

    float mi = tot.x / NG, vi = tot.y / NG - mi * mi, ri = rsqrtf(vi + LN_EPS);
    float mh = tot.z / NG, vh = tot.w / NG - mh * mh, rh = rsqrtf(vh + LN_EPS);

    float gate[4];
#pragma unroll
    for (int q = 0; q < 4; q++) {
        int col = tid + H * q;
        gate[q] = (iv[q] - mi) * ri * g_ih[col] + be_ih[col]
                + (hv[q] - mh) * rh * g_hh[col] + be_hh[col];
    }

    float cin = c[(long)b * H + tid];
    float cn  = sigmoidf_(gate[1]) * cin + sigmoidf_(gate[0]) * tanhf(gate[2]);
    float tc  = tanhf(cn);

    red[tid] = make_float4(tc, tc * tc, 0.f, 0.f);
    __syncthreads();
    for (int off = 256; off > 0; off >>= 1) {
        if (tid < off) {
            float4 a = red[tid], d = red[tid + off];
            red[tid] = make_float4(a.x + d.x, a.y + d.y, 0.f, 0.f);
        }
        __syncthreads();
    }
    float mt = red[0].x / H, vt = red[0].y / H - mt * mt, rt = rsqrtf(vt + LN_EPS);

    float hn = sigmoidf_(gate[3]) * ((tc - mt) * rt * g_ho[tid] + be_ho[tid]);
    c[(long)b * H + tid] = cn;
    h[(long)b * H + tid] = (__bf16)hn;
}

// ---------------------------------------------------------------------------
// Dense head: 512 -> 128 -> 64 -> 32 -> 1 with ReLU (tiny; scalar fp32)
// ---------------------------------------------------------------------------
__global__ void __launch_bounds__(128) head_kernel(
    const __bf16* __restrict__ h,
    const float* __restrict__ Wd1, const float* __restrict__ bd1,
    const float* __restrict__ Wd2, const float* __restrict__ bd2,
    const float* __restrict__ Wd3, const float* __restrict__ bd3,
    const float* __restrict__ Wd4, const float* __restrict__ bd4,
    float* __restrict__ out)
{
    int b = blockIdx.x, tid = threadIdx.x;
    __shared__ float hrow[512], z1[128], z2[64], z3[32];
    for (int i = tid; i < 512; i += 128) hrow[i] = (float)h[(long)b * 512 + i];
    __syncthreads();
    {
        float a = bd1[tid];
        for (int k = 0; k < 512; k++) a += hrow[k] * Wd1[tid * 512 + k];
        z1[tid] = fmaxf(a, 0.f);
    }
    __syncthreads();
    if (tid < 64) {
        float a = bd2[tid];
        for (int k = 0; k < 128; k++) a += z1[k] * Wd2[tid * 128 + k];
        z2[tid] = fmaxf(a, 0.f);
    }
    __syncthreads();
    if (tid < 32) {
        float a = bd3[tid];
        for (int k = 0; k < 64; k++) a += z2[k] * Wd3[tid * 64 + k];
        z3[tid] = fmaxf(a, 0.f);
    }
    __syncthreads();
    if (tid == 0) {
        float a = bd4[0];
        for (int k = 0; k < 32; k++) a += z3[k] * Wd4[k];
        out[b] = a;
    }
}

// ---------------------------------------------------------------------------
extern "C" void kernel_launch(void* const* d_in, const int* in_sizes, int n_in,
                              void* d_out, int out_size, void* d_ws, size_t ws_size,
                              hipStream_t stream)
{
    (void)in_sizes; (void)n_in; (void)out_size; (void)ws_size;

    const float* x      = (const float*)d_in[0];
    const float* ln1_g  = (const float*)d_in[1];
    const float* ln1_b  = (const float*)d_in[2];
    const float* W_ih0  = (const float*)d_in[3];
    const float* b_ih0  = (const float*)d_in[4];
    const float* W_hh0  = (const float*)d_in[5];
    const float* b_hh0  = (const float*)d_in[6];
    const float* g_ih0  = (const float*)d_in[7];
    const float* be_ih0 = (const float*)d_in[8];
    const float* g_hh0  = (const float*)d_in[9];
    const float* be_hh0 = (const float*)d_in[10];
    const float* g_ho0  = (const float*)d_in[11];
    const float* be_ho0 = (const float*)d_in[12];
    const float* W_ih1  = (const float*)d_in[13];
    const float* b_ih1  = (const float*)d_in[14];
    const float* W_hh1  = (const float*)d_in[15];
    const float* b_hh1  = (const float*)d_in[16];
    const float* g_ih1  = (const float*)d_in[17];
    const float* be_ih1 = (const float*)d_in[18];
    const float* g_hh1  = (const float*)d_in[19];
    const float* be_hh1 = (const float*)d_in[20];
    const float* g_ho1  = (const float*)d_in[21];
    const float* be_ho1 = (const float*)d_in[22];
    const float* Wd1 = (const float*)d_in[23];
    const float* bd1 = (const float*)d_in[24];
    const float* Wd2 = (const float*)d_in[25];
    const float* bd2 = (const float*)d_in[26];
    const float* Wd3 = (const float*)d_in[27];
    const float* bd3 = (const float*)d_in[28];
    const float* Wd4 = (const float*)d_in[29];
    const float* bd4 = (const float*)d_in[30];

    // ---- workspace carve-out (all 256B aligned) ----
    char* p = (char*)d_ws;
    auto take = [&](size_t bytes) -> char* {
        char* r = p; p += (bytes + 255) & ~(size_t)255; return r;
    };
    __bf16* xn    = (__bf16*)take(512ull * 200 * 64 * 2);   // LN'd input, bf16
    __bf16* wih0b = (__bf16*)take(2048ull * 64 * 2);
    __bf16* whh0b = (__bf16*)take(2048ull * 512 * 2);
    __bf16* wih1b = (__bf16*)take(2048ull * 512 * 2);
    __bf16* whh1b = (__bf16*)take(2048ull * 512 * 2);
    float*  Gih   = (float*)take(512ull * 2048 * 4);        // per-step ih gates
    float*  Ghh   = (float*)take(512ull * 2048 * 4);        // per-step hh gates
    char*   state = take(512ull * 512 * 2 * 2 + 512ull * 512 * 4 * 2);
    __bf16* h0 = (__bf16*)state;
    __bf16* h1 = (__bf16*)(state + 512 * 512 * 2);
    float*  c0 = (float*)(state + 512 * 512 * 4);
    float*  c1 = (float*)(state + 512 * 512 * 4 + 512 * 512 * 4);

    // zero h0,h1,c0,c1 (contiguous region; deterministic per call)
    int nz = (512 * 512 * 2 * 2 + 512 * 512 * 4 * 2) / 4;
    zero_kernel<<<(nz + 255) / 256, 256, 0, stream>>>((float*)state, nz);

    // input LN + one-shot weight conversion
    ln_input_kernel<<<512, 256, 0, stream>>>(x, ln1_g, ln1_b, xn);
    {
        long total = 2048L * 64 + 3L * 2048 * 512;
        cvt_weights_kernel<<<(int)((total + 255) / 256), 256, 0, stream>>>(
            W_ih0, W_hh0, W_ih1, W_hh1, wih0b, whh0b, wih1b, whh1b);
    }

    // 2 problems x (8 m-tiles * 32 n-tiles = 256 waves = 32 blocks) -> 64 blocks
    for (int t = 0; t < 200; t++) {
        // layer 0: ih (xn_t, K=64) and hh (h0_prev, K=512) in one launch
        gemm_dual_kernel<<<64, 256, 0, stream>>>(
            xn + (long)t * 64, 12800, wih0b, 64,  b_ih0, Gih, 64,
            h0,                512,   whh0b, 512, b_hh0, Ghh, 512);
        lstm_cell_kernel<<<512, 512, 0, stream>>>(Gih, Ghh, g_ih0, be_ih0, g_hh0, be_hh0,
                                                  g_ho0, be_ho0, c0, h0);
        // layer 1: ih (h0_new) and hh (h1_prev), both K=512, one launch
        gemm_dual_kernel<<<64, 256, 0, stream>>>(
            h0, 512, wih1b, 512, b_ih1, Gih, 512,
            h1, 512, whh1b, 512, b_hh1, Ghh, 512);
        lstm_cell_kernel<<<512, 512, 0, stream>>>(Gih, Ghh, g_ih1, be_ih1, g_hh1, be_hh1,
                                                  g_ho1, be_ho1, c1, h1);
    }

    head_kernel<<<512, 128, 0, stream>>>(h1, Wd1, bd1, Wd2, bd2, Wd3, bd3, Wd4, bd4,
                                         (float*)d_out);
}
